// DenseTrackingModule_2774548873686
// MI455X (gfx1250) — matrix operand order repaired and checked by reading, not compiled, not run on hardware
//
#include <hip/hip_runtime.h>
#include <hip/hip_bf16.h>

typedef __attribute__((ext_vector_type(16))) _Float16 v16h;
typedef __attribute__((ext_vector_type(8)))  _Float16 v8h;
typedef __attribute__((ext_vector_type(8)))  float    v8f;
typedef __attribute__((ext_vector_type(4)))  unsigned int v4u;

union ABfrag { v16h h; v4u u[2]; };

// ---------------------------------------------------------------------------
// Weight packing: OIHW fp32 -> per-wave WMMA B-fragment blocks (fp16).
// wp[((chunk*CT + ct)*32 + lane)*16 + j]
//   chunk = (r*3+s)*(CIN/32) + cc   (K-chunk of 32)
//   B-fragment K mapping (16-bit B 32x16): lane<16 -> K=0..7,16..23, lane>=16 -> K=8..15,24..31
// ---------------------------------------------------------------------------
template<int CIN, int COUT>
__global__ __launch_bounds__(256) void pack_w_kernel(const float* __restrict__ w,
                                                     _Float16* __restrict__ wp) {
    constexpr int NCC = CIN / 32;
    constexpr int CT  = COUT / 16;
    constexpr int TOTAL = 9 * NCC * CT * 512;
    int idx = blockIdx.x * 256 + threadIdx.x;
    if (idx >= TOTAL) return;
    int j    = idx & 15;
    int lane = (idx >> 4) & 31;
    int ct   = (idx >> 9) % CT;
    int chunk = idx / (512 * CT);
    int tap = chunk / NCC, cc = chunk % NCC;
    int r = tap / 3, s = tap % 3;
    int h  = lane >> 4;
    int kB = (j < 8) ? (h * 8 + j) : (16 + h * 8 + (j - 8));  // B-matrix K index within chunk
    int ci = cc * 32 + kB;
    int co = ct * 16 + (lane & 15);
    wp[idx] = (_Float16)w[((co * CIN + ci) * 3 + r) * 3 + s];
}

// ---------------------------------------------------------------------------
// conv1: 3->64, 3x3, stride 1, pad 1, fp32 math, ReLU, fp16 NHWC output.
// ---------------------------------------------------------------------------
__global__ __launch_bounds__(256) void conv1_kernel(const float* __restrict__ frames,
                                                    const float* __restrict__ w1,
                                                    const float* __restrict__ b1,
                                                    _Float16* __restrict__ act1) {
    __shared__ float ws[64 * 27];
    __shared__ float sb[64];
    for (int i = threadIdx.x; i < 64 * 27; i += 256) ws[i] = w1[i];
    if (threadIdx.x < 64) sb[threadIdx.x] = b1[threadIdx.x];
    __syncthreads();

    int pix = blockIdx.x * 256 + threadIdx.x;       // (n*512 + oh)*512 + ow
    int ow = pix & 511;
    int oh = (pix >> 9) & 511;
    int n  = pix >> 18;

    float x[27];
#pragma unroll
    for (int c = 0; c < 3; ++c)
#pragma unroll
        for (int r = 0; r < 3; ++r) {
            int ih = oh + r - 1;
#pragma unroll
            for (int s = 0; s < 3; ++s) {
                int iw = ow + s - 1;
                bool ok = ((unsigned)ih < 512u) && ((unsigned)iw < 512u);
                int ihc = ok ? ih : 0, iwc = ok ? iw : 0;
                float v = frames[(((size_t)n * 3 + c) * 512 + ihc) * 512 + iwc];
                x[c * 9 + r * 3 + s] = ok ? v : 0.0f;
            }
        }

    _Float16* outp = act1 + (size_t)pix * 64;
    for (int co = 0; co < 64; ++co) {
        float acc = sb[co];
#pragma unroll
        for (int i = 0; i < 27; ++i) acc = fmaf(ws[co * 27 + i], x[i], acc);
        outp[co] = (_Float16)(acc > 0.0f ? acc : 0.0f);
    }
}

// ---------------------------------------------------------------------------
// Implicit-GEMM 3x3 stride-2 pad-1 conv, WMMA f32_16x16x32_f16.
// One wave = 64 pixels x 64 couts: 4 A-fragments x 4 B-fragments -> 16 WMMAs
// per K-chunk (1.0 b128 load per WMMA).
// OOB: only M-tile 0 can be OOB per-lane (left edge); tiles 1-3 only need the
// wave-uniform top-row mask. All handled by clamp + select, no EXEC branches.
// ---------------------------------------------------------------------------
template<int CIN, int HIN, int HOUT, int COUT, bool OUT_FP32>
__global__ __launch_bounds__(256) void conv_wmma_kernel(const _Float16* __restrict__ in,
                                                        const _Float16* __restrict__ wpack,
                                                        const float* __restrict__ bias,
                                                        void* __restrict__ outp) {
    constexpr int NCC = CIN / 32;
    constexpr int WT  = HOUT / 64;       // 64-pixel M super-tiles
    constexpr int CT  = COUT / 16;
    constexpr int CG  = CT / 4;          // cout groups of 64 channels

    const int lane = threadIdx.x & 31;
    const int col  = lane & 15;
    const int hi   = lane >> 4;

    unsigned g = blockIdx.x * 8u + (threadIdx.x >> 5);
    const int cg = g % CG;  g /= CG;
    const int wt = g % WT;  g /= WT;
    const int oh = g % HOUT; g /= HOUT;
    const int n  = g;

    const int ow0 = wt * 64 + col;       // A-row pixel, M-tile 0 (tiles at +16t)

    v8f acc[4][4];                       // [m-tile][cout-subtile]
#pragma unroll
    for (int t = 0; t < 4; ++t)
#pragma unroll
        for (int u = 0; u < 4; ++u) acc[t][u] = (v8f){};

    const _Float16* wb = wpack + (size_t)(cg * 4) * 512 + (size_t)lane * 16;
    const v4u zz = {0u, 0u, 0u, 0u};

#pragma unroll
    for (int r = 0; r < 3; ++r) {
        const int ih = oh * 2 - 1 + r;
        const bool hok = (unsigned)ih < (unsigned)HIN;
        const int ihc = hok ? ih : 0;
        const _Float16* rowp = in + ((size_t)n * HIN + ihc) * HIN * CIN;  // wave-uniform
        __builtin_prefetch(rowp + (size_t)(wt * 128) * CIN, 0, 3);
#pragma unroll
        for (int s = 0; s < 3; ++s) {
            const int iw0 = ow0 * 2 - 1 + s;
            // tile t reads iw0 + 32*t ; only tile 0 can be OOB horizontally
            const bool ok0 = hok && ((unsigned)iw0 < (unsigned)HIN);
            const int iwc0 = ((unsigned)iw0 < (unsigned)HIN) ? iw0 : 0;
            int aoff[4];
            aoff[0] = iwc0 * CIN + hi * 8;
            aoff[1] = (iw0 + 32) * CIN + hi * 8;
            aoff[2] = (iw0 + 64) * CIN + hi * 8;
            aoff[3] = (iw0 + 96) * CIN + hi * 8;
#pragma unroll
            for (int cc = 0; cc < NCC; ++cc) {
                const int c0 = cc * 32;
                ABfrag a[4];
#pragma unroll
                for (int t = 0; t < 4; ++t) {
                    v4u lo = *(const v4u*)(rowp + aoff[t] + c0);
                    v4u hh = *(const v4u*)(rowp + aoff[t] + c0 + 16);
                    const bool ok = (t == 0) ? ok0 : hok;
                    lo = ok ? lo : zz;
                    hh = ok ? hh : zz;
                    a[t].u[0] = lo;
                    a[t].u[1] = hh;
                }

                const int chunk = (r * 3 + s) * NCC + cc;
                const _Float16* bp = wb + (size_t)chunk * (CT * 512);
                v16h b[4];
#pragma unroll
                for (int u = 0; u < 4; ++u) b[u] = *(const v16h*)(bp + u * 512);

#pragma unroll
                for (int u = 0; u < 4; ++u)
#pragma unroll
                    for (int t = 0; t < 4; ++t)
                        acc[t][u] = __builtin_amdgcn_wmma_f32_16x16x32_f16(
                            false, a[t].h, false, b[u], (short)0, acc[t][u], false, false);
            }
        }
    }

    // Epilogue: D layout (32-bit C/D 16x16): lane = col N, VGPR v -> row m = hi*8 + v
    const size_t base = ((size_t)n * HOUT + oh) * HOUT;
#pragma unroll
    for (int u = 0; u < 4; ++u) {
        const int ct = cg * 4 + u;
        const float bv = bias[ct * 16 + col];
#pragma unroll
        for (int t = 0; t < 4; ++t) {
#pragma unroll
            for (int v = 0; v < 8; ++v) {
                const int m  = hi * 8 + v;
                const int ow = wt * 64 + t * 16 + m;
                float val = acc[t][u][v] + bv;
                val = val > 0.0f ? val : 0.0f;
                const size_t o = (base + ow) * COUT + ct * 16 + col;
                if (OUT_FP32) ((float*)outp)[o] = val;
                else          ((_Float16*)outp)[o] = (_Float16)val;
            }
        }
    }
}

// ---------------------------------------------------------------------------
// Cosine similarity over 256 channels + mean over H + broadcast.
// One block per (t,b,w), 128 threads over h. feats: fp32 NHWC [16][128][128][256].
// ---------------------------------------------------------------------------
__global__ __launch_bounds__(128) void cosine_kernel(const float* __restrict__ feats,
                                                     float* __restrict__ out) {
    __shared__ float sd[128];
    int idx = blockIdx.x;
    int w = idx & 127; idx >>= 7;
    int b = idx & 3;
    int t = idx >> 2;
    int h = threadIdx.x;

    const float* f0 = feats + (((size_t)(t * 4 + b) * 128 + h) * 128 + w) * 256;
    const float* f1 = feats + (((size_t)((t + 1) * 4 + b) * 128 + h) * 128 + w) * 256;

    float num = 0.0f, n0 = 0.0f, n1 = 0.0f;
#pragma unroll 4
    for (int c = 0; c < 256; c += 4) {
        float4 a  = *(const float4*)(f0 + c);
        float4 bb = *(const float4*)(f1 + c);
        num += a.x * bb.x + a.y * bb.y + a.z * bb.z + a.w * bb.w;
        n0  += a.x * a.x + a.y * a.y + a.z * a.z + a.w * a.w;
        n1  += bb.x * bb.x + bb.y * bb.y + bb.z * bb.z + bb.w * bb.w;
    }
    float d0 = sqrtf(n0); d0 = d0 > 1e-8f ? d0 : 1e-8f;
    float d1 = sqrtf(n1); d1 = d1 > 1e-8f ? d1 : 1e-8f;
    sd[h] = num / (d0 * d1);
    __syncthreads();
    for (int off = 64; off > 0; off >>= 1) {
        if (h < off) sd[h] += sd[h + off];
        __syncthreads();
    }
    float mean = sd[0] * (1.0f / 128.0f);
    out[((size_t)(t * 4 + b) * 128 + h) * 128 + w] = mean;
}

// ---------------------------------------------------------------------------
extern "C" void kernel_launch(void* const* d_in, const int* in_sizes, int n_in,
                              void* d_out, int out_size, void* d_ws, size_t ws_size,
                              hipStream_t stream) {
    const float* frames = (const float*)d_in[0];
    const float* w1 = (const float*)d_in[1];
    const float* b1 = (const float*)d_in[2];
    const float* w2 = (const float*)d_in[3];
    const float* b2 = (const float*)d_in[4];
    const float* w3 = (const float*)d_in[5];
    const float* b3 = (const float*)d_in[6];
    float* out = (float*)d_out;

    char* ws = (char*)d_ws;
    size_t off = 0;
    _Float16* wp2  = (_Float16*)(ws + off); off += (size_t)18 * 8 * 512 * 2;            // 144 KB
    _Float16* wp3  = (_Float16*)(ws + off); off += (size_t)36 * 16 * 512 * 2;           // 576 KB
    _Float16* act1 = (_Float16*)(ws + off); off += (size_t)16 * 512 * 512 * 64 * 2;     // 512 MB
    _Float16* act2 = (_Float16*)(ws + off); off += (size_t)16 * 256 * 256 * 128 * 2;    // 256 MB
    float*    feats = (float*)(ws + off);   off += (size_t)16 * 128 * 128 * 256 * 4;    // 256 MB
    (void)ws_size; (void)in_sizes; (void)n_in; (void)out_size;

    pack_w_kernel<64, 128><<<288, 256, 0, stream>>>(w2, wp2);
    pack_w_kernel<128, 256><<<1152, 256, 0, stream>>>(w3, wp3);

    conv1_kernel<<<16384, 256, 0, stream>>>(frames, w1, b1, act1);

    // conv2: waves = 16(n)*256(oh)*4(wt)*2(cg) = 32768 -> 4096 blocks of 8 waves
    conv_wmma_kernel<64, 512, 256, 128, false><<<4096, 256, 0, stream>>>(act1, wp2, b2, act2);

    // conv3: waves = 16*128*2(wt)*4(cg) = 16384 -> 2048 blocks
    conv_wmma_kernel<128, 256, 128, 256, true><<<2048, 256, 0, stream>>>(act2, wp3, b3, feats);

    cosine_kernel<<<1536, 128, 0, stream>>>(feats, out);
}
